// TemporalMoEETA_2894807957598
// MI455X (gfx1250) — matrix operand order repaired and checked by reading, not compiled, not run on hardware
//
#include <hip/hip_runtime.h>
#include <hip/hip_bf16.h>

// ---------------------------------------------------------------------------
// TemporalMoEETA for MI455X (gfx1250): WMMA f16 GEMMs (packed-f16 weights),
// fused GRU (async-LDS table staging), fused MoE.
// ---------------------------------------------------------------------------

typedef __attribute__((ext_vector_type(16))) _Float16 v16h;
typedef __attribute__((ext_vector_type(8)))  float    v8f;

#define E_STATIC_C  300000

// ---------------- device helpers ----------------

__device__ __forceinline__ v8f wmma16(v16h a, v16h b, v8f c) {
  return __builtin_amdgcn_wmma_f32_16x16x32_f16(false, a, false, b, (short)0, c,
                                                false, false);
}

__device__ __forceinline__ v8f splat8(float v) {
  v8f a;
#pragma unroll
  for (int r = 0; r < 8; ++r) a[r] = v;
  return a;
}

__device__ __forceinline__ float gelu_f(float x) {
  float x3 = x * x * x;
  return 0.5f * x * (1.f + tanhf(0.7978845608028654f * (x + 0.044715f * x3)));
}

__device__ __forceinline__ float sigm_f(float x) { return 1.f / (1.f + expf(-x)); }

__device__ __forceinline__ unsigned f2key(float f) {
  unsigned b = __float_as_uint(f);
  return (b & 0x80000000u) ? ~b : (b | 0x80000000u);
}
__device__ __forceinline__ float key2f(unsigned k) {
  return __uint_as_float((k & 0x80000000u) ? (k ^ 0x80000000u) : ~k);
}

__device__ __forceinline__ v16h cvt16(float4 q0, float4 q1, float4 q2,
                                      float4 q3) {
  v16h a;
  a[0] = (_Float16)q0.x;  a[1] = (_Float16)q0.y;
  a[2] = (_Float16)q0.z;  a[3] = (_Float16)q0.w;
  a[4] = (_Float16)q1.x;  a[5] = (_Float16)q1.y;
  a[6] = (_Float16)q1.z;  a[7] = (_Float16)q1.w;
  a[8] = (_Float16)q2.x;  a[9] = (_Float16)q2.y;
  a[10] = (_Float16)q2.z; a[11] = (_Float16)q2.w;
  a[12] = (_Float16)q3.x; a[13] = (_Float16)q3.y;
  a[14] = (_Float16)q3.z; a[15] = (_Float16)q3.w;
  return a;
}

// A fragment (16x32 f16). Requires: rows row0..row0+15 valid, K chunk full,
// (row*lda + k0) a multiple of 8. elem i<8 -> K=kb+i ; i>=8 -> K=16+kb+(i-8).
__device__ __forceinline__ v16h load_a_full(const float* A, int lda, int row0,
                                            int k0) {
  int lane = threadIdx.x & 31;
  const float* p =
      A + (size_t)(row0 + (lane & 15)) * lda + k0 + ((lane >> 4) << 3);
  const float4* p4 = (const float4*)p;
  return cvt16(p4[0], p4[1], p4[4], p4[5]);
}

// A fragment for K==16 (k0==0): second K-half statically zero.
__device__ __forceinline__ v16h load_a_k16(const float* A, int lda, int row0) {
  int lane = threadIdx.x & 31;
  const float* p = A + (size_t)(row0 + (lane & 15)) * lda + ((lane >> 4) << 3);
  const float4* p4 = (const float4*)p;
  float4 q0 = p4[0], q1 = p4[1];
  v16h a;
  a[0] = (_Float16)q0.x; a[1] = (_Float16)q0.y;
  a[2] = (_Float16)q0.z; a[3] = (_Float16)q0.w;
  a[4] = (_Float16)q1.x; a[5] = (_Float16)q1.y;
  a[6] = (_Float16)q1.z; a[7] = (_Float16)q1.w;
#pragma unroll
  for (int i = 8; i < 16; ++i) a[i] = (_Float16)0.f;
  return a;
}

// A fragment for GRU edge_seq slice: lda=21, K=7 (branchless cndmask zeroing).
__device__ __forceinline__ v16h load_a_k7(const float* base21, int row0) {
  int lane = threadIdx.x & 31;
  int kb = (lane >> 4) << 3;
  const float* p = base21 + (size_t)(row0 + (lane & 15)) * 21;
  float f[7];
#pragma unroll
  for (int i = 0; i < 7; ++i) f[i] = p[i];  // unconditional, always in-bounds
  v16h a;
#pragma unroll
  for (int i = 0; i < 8; ++i)
    a[i] = (_Float16)((kb == 0 && i < 7) ? f[i < 7 ? i : 0] : 0.f);
#pragma unroll
  for (int i = 8; i < 16; ++i) a[i] = (_Float16)0.f;
  return a;
}

// A fragment from LDS staging buffer [16][128] f32 (ds_load_b128 x4)
__device__ __forceinline__ v16h load_a_lds(const float* buf, int k0) {
  int lane = threadIdx.x & 31;
  const float* p = buf + (lane & 15) * 128 + k0 + ((lane >> 4) << 3);
  const float4* p4 = (const float4*)p;
  return cvt16(p4[0], p4[1], p4[4], p4[5]);
}

// Packed-f16 B fragment: BP layout [chunk][N][32]; elem i<8 -> K=kb+i, i>=8 -> K=16+kb+(i-8).
__device__ __forceinline__ v16h load_b_packed(const _Float16* BP, int N,
                                              int chunk, int col0) {
  int lane = threadIdx.x & 31;
  const _Float16* p =
      BP + ((size_t)chunk * N + col0 + (lane & 15)) * 32 + ((lane >> 4) << 3);
  v16h b;
#pragma unroll
  for (int i = 0; i < 8; ++i) { b[i] = p[i]; b[8 + i] = p[16 + i]; }
  return b;
}

// Packed-f16 GRU weight (layout identical to W[384][128], just f16):
// B[k][col] = WP[col*128 + k]; per lane halves kb..kb+7 and 16+kb..16+kb+7.
__device__ __forceinline__ v16h load_w_packed(const _Float16* WP, int col,
                                              int k0) {
  int lane = threadIdx.x & 31;
  const _Float16* p = WP + (size_t)col * 128 + k0 + ((lane >> 4) << 3);
  v16h b;
#pragma unroll
  for (int i = 0; i < 8; ++i) { b[i] = p[i]; b[8 + i] = p[16 + i]; }
  return b;
}

// CDNA5 async global->LDS copy of one 32-bit element (ASYNCcnt-tracked).
__device__ __forceinline__ void async_g2l_b32(unsigned lds_off,
                                              const float* gptr) {
  unsigned long long ga = (unsigned long long)(size_t)gptr;
  asm volatile("global_load_async_to_lds_b32 %0, %1, off" ::"v"(lds_off),
               "v"(ga)
               : "memory");
}

// ---------------- utility kernels ----------------

__global__ void k_fill_u32(unsigned* p, unsigned v, long n) {
  long i = (long)blockIdx.x * blockDim.x + threadIdx.x;
  if (i < n) p[i] = v;
}

// pack row-major f32 B[K][N] -> f16 [chunk][N][32] (zero-padded K tail)
__global__ void k_packB(const float* B, int K, int N, _Float16* out) {
  long total = (long)((K + 31) >> 5) * N * 32;
  long i = (long)blockIdx.x * blockDim.x + threadIdx.x;
  if (i >= total) return;
  int kk = (int)(i & 31);
  long rem = i >> 5;
  int n = (int)(rem % N);
  int chunk = (int)(rem / N);
  int k = chunk * 32 + kk;
  float v = (k < K) ? B[(size_t)k * N + n] : 0.f;
  out[i] = (_Float16)v;
}

__global__ void k_packH(const float* W, _Float16* out, long n) {
  long i = (long)blockIdx.x * blockDim.x + threadIdx.x;
  if (i < n) out[i] = (_Float16)W[i];
}

__global__ void k_ea_mean(const float* eattr, float* ea_sum, int nE) {
  __shared__ float s7[7];
  if (threadIdx.x < 7) s7[threadIdx.x] = 0.f;
  __syncthreads();
  long i = (long)blockIdx.x * blockDim.x + threadIdx.x;
  if (i < nE) {
#pragma unroll
    for (int o = 0; o < 7; ++o) atomicAdd(&s7[o], eattr[i * 7 + o]);
  }
  __syncthreads();
  if (threadIdx.x < 7) atomicAdd(&ea_sum[threadIdx.x], s7[threadIdx.x]);
}

// ------------- generic WMMA GEMM: C = A@B + bias (packed-f16 B) -------------
// Requires M%16==0, N%32==0, K%32==0 or K==16.
__global__ __launch_bounds__(256) void k_gemm(const float* A, int lda,
                                              const _Float16* BP,
                                              const float* bias, float* C,
                                              int ldc, int M, int N, int K) {
  int wave = threadIdx.x >> 5, lane = threadIdx.x & 31;
  int nt2 = N >> 5;  // 32-wide column tiles
  long tile = (long)blockIdx.x * 8 + wave;
  long mt = tile / nt2;
  int nt = (int)(tile % nt2);
  int row0 = (int)mt * 16, col0 = nt * 32;
  if (row0 >= M) return;
  int c0 = col0 + (lane & 15);
  v8f acc0 = splat8(bias[c0]);
  v8f acc1 = splat8(bias[c0 + 16]);
  if (K == 16) {
    v16h a = load_a_k16(A, lda, row0);
    acc0 = wmma16(a, load_b_packed(BP, N, 0, col0), acc0);
    acc1 = wmma16(a, load_b_packed(BP, N, 0, col0 + 16), acc1);
  } else {
    int chunks = K >> 5;
    for (int ch = 0; ch < chunks; ++ch) {
      v16h a = load_a_full(A, lda, row0, ch * 32);
      acc0 = wmma16(a, load_b_packed(BP, N, ch, col0), acc0);
      acc1 = wmma16(a, load_b_packed(BP, N, ch, col0 + 16), acc1);
    }
  }
  int rbase = row0 + ((lane >> 4) << 3);
#pragma unroll
  for (int r = 0; r < 8; ++r) {
    C[(size_t)(rbase + r) * ldc + c0] = acc0[r];
    C[(size_t)(rbase + r) * ldc + c0 + 16] = acc1[r];
  }
}

// ---------------- GAT edge passes (wave per edge) ----------------

__global__ __launch_bounds__(128) void k_edge_logits(
    const int* ei, const float* eattr, const float* ea_sum, const float* xl,
    const float* xr, const float* We, const float* att, float* logits,
    unsigned* mx, int nE, int nN) {
  __shared__ float sWe[7 * 128];
  __shared__ float satt[128];
  for (int i = threadIdx.x; i < 7 * 128; i += 128) sWe[i] = We[i];
  if (threadIdx.x < 128) satt[threadIdx.x] = att[threadIdx.x];
  __syncthreads();
  int wave = threadIdx.x >> 5, lane = threadIdx.x & 31;
  long e = (long)blockIdx.x * 4 + wave;
  long tot = (long)nE + nN;
  if (e >= tot) return;
  int src, dst;
  float ea[7];
  if (e < nE) {
    src = ei[e];
    dst = ei[(long)nE + e];
#pragma unroll
    for (int o = 0; o < 7; ++o) ea[o] = eattr[e * 7 + o];
  } else {
    src = dst = (int)(e - nE);
    float inv = 1.f / (float)nE;
#pragma unroll
    for (int o = 0; o < 7; ++o) ea[o] = ea_sum[o] * inv;
  }
  float4 xlv = ((const float4*)xl)[(size_t)src * 32 + lane];
  float4 xrv = ((const float4*)xr)[(size_t)dst * 32 + lane];
  float xla[4] = {xlv.x, xlv.y, xlv.z, xlv.w};
  float xra[4] = {xrv.x, xrv.y, xrv.z, xrv.w};
  float l0 = 0.f;
#pragma unroll
  for (int q = 0; q < 4; ++q) {
    int d = lane * 4 + q;
    float ew = 0.f;
#pragma unroll
    for (int o = 0; o < 7; ++o) ew += ea[o] * sWe[o * 128 + d];
    float m = xla[q] + xra[q] + ew;
    m = (m > 0.f) ? m : 0.2f * m;  // leaky_relu 0.2
    l0 += m * satt[d];
  }
  l0 += __shfl_xor(l0, 1);
  l0 += __shfl_xor(l0, 2);
  l0 += __shfl_xor(l0, 4);
  if ((lane & 7) == 0) {
    int h = lane >> 3;
    logits[e * 4 + h] = l0;
    atomicMax(&mx[(size_t)dst * 4 + h], f2key(l0));
  }
}

__global__ void k_edge_soft(const int* ei, const float* logits,
                            const unsigned* mx, float* exb, float* den, int nE,
                            int nN) {
  long i = (long)blockIdx.x * blockDim.x + threadIdx.x;
  long tot = ((long)nE + nN) * 4;
  if (i >= tot) return;
  long e = i >> 2;
  int h = (int)(i & 3);
  int dst = (e < nE) ? ei[(long)nE + e] : (int)(e - nE);
  float m = key2f(mx[(size_t)dst * 4 + h]);
  float v = expf(logits[i] - m);
  exb[i] = v;
  atomicAdd(&den[(size_t)dst * 4 + h], v);
}

__global__ __launch_bounds__(128) void k_edge_agg(const int* ei,
                                                  const float* exb,
                                                  const float* den,
                                                  const float* xl, float* acc,
                                                  int nE, int nN) {
  int wave = threadIdx.x >> 5, lane = threadIdx.x & 31;
  long e = (long)blockIdx.x * 4 + wave;
  long tot = (long)nE + nN;
  if (e >= tot) return;
  int src, dst;
  if (e < nE) {
    src = ei[e];
    dst = ei[(long)nE + e];
  } else {
    src = dst = (int)(e - nE);
  }
  int h = lane >> 3;
  float alpha = exb[e * 4 + h] / den[(size_t)dst * 4 + h];
  float4 xv = ((const float4*)xl)[(size_t)src * 32 + lane];
  float* ap = acc + (size_t)dst * 128 + lane * 4;
  atomicAdd(ap + 0, alpha * xv.x);
  atomicAdd(ap + 1, alpha * xv.y);
  atomicAdd(ap + 2, alpha * xv.z);
  atomicAdd(ap + 3, alpha * xv.w);
}

// ---------------- node updates (wave per node) ----------------

__global__ __launch_bounds__(128) void k_node1(const float* acc,
                                               const float* b_out,
                                               const float* x,
                                               const float* skW,
                                               const float* skb,
                                               const float* lng,
                                               const float* lnb, float* hout,
                                               int nN) {
  __shared__ float sW[16 * 128];
  for (int i = threadIdx.x; i < 16 * 128; i += 128) sW[i] = skW[i];
  __syncthreads();
  int wave = threadIdx.x >> 5, lane = threadIdx.x & 31;
  long n = (long)blockIdx.x * 4 + wave;
  if (n >= nN) return;
  float xr_[16];
#pragma unroll
  for (int o = 0; o < 16; ++o) xr_[o] = x[n * 16 + o];
  float vals[4], s1 = 0.f, s2 = 0.f;
#pragma unroll
  for (int q = 0; q < 4; ++q) {
    int d = lane + 32 * q;
    float g = gelu_f(acc[n * 128 + d] + b_out[d]);
    float sk = skb[d];
#pragma unroll
    for (int o = 0; o < 16; ++o) sk += xr_[o] * sW[o * 128 + d];
    float t = g + sk;
    vals[q] = t;
    s1 += t;
    s2 += t * t;
  }
#pragma unroll
  for (int m = 16; m; m >>= 1) {
    s1 += __shfl_xor(s1, m);
    s2 += __shfl_xor(s2, m);
  }
  float mu = s1 * (1.f / 128.f);
  float inv = rsqrtf(s2 * (1.f / 128.f) - mu * mu + 1e-5f);
#pragma unroll
  for (int q = 0; q < 4; ++q) {
    int d = lane + 32 * q;
    hout[n * 128 + d] = lng[d] * (vals[q] - mu) * inv + lnb[d];
  }
}

__global__ __launch_bounds__(128) void k_node2(const float* acc,
                                               const float* b_out,
                                               const float* lng,
                                               const float* lnb, float* h,
                                               int nN) {
  int wave = threadIdx.x >> 5, lane = threadIdx.x & 31;
  long n = (long)blockIdx.x * 4 + wave;
  if (n >= nN) return;
  float vals[4], s1 = 0.f, s2 = 0.f;
#pragma unroll
  for (int q = 0; q < 4; ++q) {
    int d = lane + 32 * q;
    float t = gelu_f(acc[n * 128 + d] + b_out[d]) + h[n * 128 + d];
    vals[q] = t;
    s1 += t;
    s2 += t * t;
  }
#pragma unroll
  for (int m = 16; m; m >>= 1) {
    s1 += __shfl_xor(s1, m);
    s2 += __shfl_xor(s2, m);
  }
  float mu = s1 * (1.f / 128.f);
  float inv = rsqrtf(s2 * (1.f / 128.f) - mu * mu + 1e-5f);
#pragma unroll
  for (int q = 0; q < 4; ++q) {
    int d = lane + 32 * q;
    h[n * 128 + d] = lng[d] * (vals[q] - mu) * inv + lnb[d];
  }
}

// ------- fused GRU + edge_to_context (wave per 16-edge tile, f16 weights) ----

__global__ __launch_bounds__(128) void k_gru(
    const float* es, const _Float16* WinP, const float* bin,
    const _Float16* WihP, const float* bih, const _Float16* WhhP,
    const float* bhh, const float* Wout, const float* bout, const float* cW,
    const float* cb, const float* lng, const float* lnb, float* ctx_accum,
    int E) {
  __shared__ float sbin[128];
  __shared__ float sbih[384];
  __shared__ float sbhh[384];
  __shared__ float sWout[128 * 7];
  __shared__ float sbout[8];
  __shared__ float scW[7 * 128];
  __shared__ float scb[128], slng[128], slnb[128];
  __shared__ float stage[4][16 * 128];
  __shared__ float te_s[4][16 * 8];
  __shared__ float ctxsum[128];

  // ASYNCcnt-tracked global->LDS staging of the two 7x128 tables; the
  // remaining small tables load through VGPRs meanwhile.
  {
    unsigned lW = (unsigned)(size_t)(void*)scW;
    unsigned lO = (unsigned)(size_t)(void*)sWout;
    for (int i = threadIdx.x; i < 7 * 128; i += 128) {
      async_g2l_b32(lW + (unsigned)i * 4u, cW + i);
      async_g2l_b32(lO + (unsigned)i * 4u, Wout + i);
    }
  }
  for (int i = threadIdx.x; i < 384; i += 128) {
    sbih[i] = bih[i];
    sbhh[i] = bhh[i];
  }
  {
    int i = threadIdx.x;
    sbin[i] = bin[i];
    scb[i] = cb[i];
    slng[i] = lng[i];
    slnb[i] = lnb[i];
    ctxsum[i] = 0.f;
  }
  if (threadIdx.x < 7) sbout[threadIdx.x] = bout[threadIdx.x];
  asm volatile("s_wait_asynccnt 0" ::: "memory");
  __syncthreads();

  int wave = threadIdx.x >> 5, lane = threadIdx.x & 31;
  int e0 = ((int)blockIdx.x * 4 + wave) * 16;
  float* buf = stage[wave];
  const int crow = lane & 15;
  const int half = lane >> 4;

  if (e0 < E) {
    if (e0 + 16 < E)  // prefetch next edge tile (speculative)
      __builtin_prefetch(es + (size_t)(e0 + 16) * 21 + lane * 10, 0, 1);

    v8f hc[8];
#pragma unroll
    for (int j = 0; j < 8; ++j) hc[j] = splat8(0.f);

#pragma unroll
    for (int t = 0; t < 3; ++t) {
      v16h hfr[4];
      if (t == 0) {
#pragma unroll
        for (int k = 0; k < 4; ++k) {
          v16h z;
#pragma unroll
          for (int i = 0; i < 16; ++i) z[i] = (_Float16)0.f;
          hfr[k] = z;
        }
      } else {
#pragma unroll
        for (int j = 0; j < 8; ++j) {
#pragma unroll
          for (int r = 0; r < 8; ++r)
            buf[(half * 8 + r) * 128 + j * 16 + crow] = hc[j][r];
        }
        asm volatile("s_wait_dscnt 0" ::: "memory");
#pragma unroll
        for (int k = 0; k < 4; ++k) hfr[k] = load_a_lds(buf, k * 32);
      }

      // xt = es[:,t,:] @ Win + bin  (K=7 padded; Win packed w/ zero tail)
      v16h esfr = load_a_k7(es + t * 7, e0);
      v8f xt[8];
#pragma unroll
      for (int j = 0; j < 8; ++j) {
        v8f a = splat8(sbin[j * 16 + crow]);
        xt[j] = wmma16(esfr, load_b_packed(WinP, 128, 0, j * 16), a);
      }
#pragma unroll
      for (int j = 0; j < 8; ++j) {
#pragma unroll
        for (int r = 0; r < 8; ++r)
          buf[(half * 8 + r) * 128 + j * 16 + crow] = xt[j][r];
      }
      asm volatile("s_wait_dscnt 0" ::: "memory");
      v16h xfr[4];
#pragma unroll
      for (int k = 0; k < 4; ++k) xfr[k] = load_a_lds(buf, k * 32);

      // gates: gi = xt@Wih^T, gh = h@Whh^T ; split r|z|n
#pragma unroll
      for (int j = 0; j < 8; ++j) {
        int c_r = j * 16 + crow;
        int c_z = 128 + j * 16 + crow;
        int c_n = 256 + j * 16 + crow;
        v8f ir = splat8(sbih[c_r]), hr = splat8(sbhh[c_r]);
        v8f iz = splat8(sbih[c_z]), hz = splat8(sbhh[c_z]);
        v8f in_ = splat8(sbih[c_n]), hn = splat8(sbhh[c_n]);
#pragma unroll
        for (int k = 0; k < 4; ++k) {
          ir = wmma16(xfr[k], load_w_packed(WihP, c_r, k * 32), ir);
          hr = wmma16(hfr[k], load_w_packed(WhhP, c_r, k * 32), hr);
          iz = wmma16(xfr[k], load_w_packed(WihP, c_z, k * 32), iz);
          hz = wmma16(hfr[k], load_w_packed(WhhP, c_z, k * 32), hz);
          in_ = wmma16(xfr[k], load_w_packed(WihP, c_n, k * 32), in_);
          hn = wmma16(hfr[k], load_w_packed(WhhP, c_n, k * 32), hn);
        }
#pragma unroll
        for (int r = 0; r < 8; ++r) {
          float rr = sigm_f(ir[r] + hr[r]);
          float zz = sigm_f(iz[r] + hz[r]);
          float nv = tanhf(in_[r] + rr * hn[r]);
          hc[j][r] = (1.f - zz) * nv + zz * hc[j][r];
        }
      }
    }  // t

    // stage hT, project to 7 dims, edge_to_context + LN + accumulate
#pragma unroll
    for (int j = 0; j < 8; ++j) {
#pragma unroll
      for (int r = 0; r < 8; ++r)
        buf[(half * 8 + r) * 128 + j * 16 + crow] = hc[j][r];
    }
    asm volatile("s_wait_dscnt 0" ::: "memory");

    float* te = te_s[wave];
    for (int task = lane; task < 112; task += 32) {
      int r = task / 7, o = task - 7 * r;
      if (e0 + r < E) {
        float s = sbout[o];
        for (int k = 0; k < 128; ++k) s += buf[r * 128 + k] * sWout[k * 7 + o];
        te[r * 8 + o] = s;
      }
    }
    asm volatile("s_wait_dscnt 0" ::: "memory");

    for (int r = 0; r < 16; ++r) {
      if (e0 + r >= E) break;  // uniform across wave (E%16==0 anyway)
      float vals[4], s1 = 0.f, s2 = 0.f;
#pragma unroll
      for (int q = 0; q < 4; ++q) {
        int d = lane + 32 * q;
        float a = scb[d];
#pragma unroll
        for (int o = 0; o < 7; ++o) a += te[r * 8 + o] * scW[o * 128 + d];
        a = gelu_f(a);
        vals[q] = a;
        s1 += a;
        s2 += a * a;
      }
#pragma unroll
      for (int m = 16; m; m >>= 1) {
        s1 += __shfl_xor(s1, m);
        s2 += __shfl_xor(s2, m);
      }
      float mu = s1 * (1.f / 128.f);
      float inv = rsqrtf(s2 * (1.f / 128.f) - mu * mu + 1e-5f);
#pragma unroll
      for (int q = 0; q < 4; ++q) {
        int d = lane + 32 * q;
        atomicAdd(&ctxsum[d], slng[d] * (vals[q] - mu) * inv + slnb[d]);
      }
    }
  }  // e0 < E

  __syncthreads();
  atomicAdd(&ctx_accum[threadIdx.x], ctxsum[threadIdx.x]);
}

// ---------------- route branch ----------------

__global__ __launch_bounds__(1024) void k_scan(const int* splits, int* offs,
                                               int n) {
  __shared__ int s[1024];
  __shared__ int carry;
  if (threadIdx.x == 0) carry = 0;
  __syncthreads();
  for (int base = 0; base < n; base += 1024) {
    int i = base + threadIdx.x;
    int v = (i < n) ? splits[i] : 0;
    s[threadIdx.x] = v;
    __syncthreads();
    for (int off = 1; off < 1024; off <<= 1) {
      int t = (threadIdx.x >= off) ? s[threadIdx.x - off] : 0;
      __syncthreads();
      s[threadIdx.x] += t;
      __syncthreads();
    }
    if (i < n) offs[i] = carry + s[threadIdx.x] - v;  // exclusive
    __syncthreads();
    if (threadIdx.x == 1023) carry += s[1023];
    __syncthreads();
  }
}

__global__ __launch_bounds__(128) void k_route_pool(const float* emb,
                                                    const int* rflat,
                                                    const int* offs,
                                                    const int* splits,
                                                    float* route_z, int nv) {
  int wave = threadIdx.x >> 5, lane = threadIdx.x & 31;
  long v = (long)blockIdx.x * 4 + wave;
  if (v >= nv) return;
  int len = splits[v];
  int off = offs[v];
  float a0 = 0.f, a1 = 0.f;
  for (int j = 0; j < len; ++j) {
    int id = rflat[off + j];
    const float* er = emb + (size_t)id * 64;
    a0 += er[lane];
    a1 += er[lane + 32];
  }
  float inv = 1.f / (float)((len > 0) ? len : 1);
  route_z[v * 64 + lane] = a0 * inv;
  route_z[v * 64 + lane + 32] = a1 * inv;
}

// ---------------- fusion head ----------------

__global__ void k_z(const float* h, const int* veh, const float* ctx,
                    const float* rz, float* z, int nv) {
  long i = (long)blockIdx.x * blockDim.x + threadIdx.x;
  if (i >= (long)nv * 320) return;
  long v = i / 320;
  int d = (int)(i - v * 320);
  float val;
  if (d < 128)
    val = h[(size_t)veh[v] * 128 + d];
  else if (d < 256)
    val = ctx[d - 128] * (1.f / (float)E_STATIC_C);
  else
    val = rz[v * 64 + (d - 256)];
  z[i] = val;
}

__global__ __launch_bounds__(256) void k_geluln256(float* buf, const float* g,
                                                   const float* b) {
  __shared__ float red[16];
  __shared__ float stats[2];
  long r = blockIdx.x;
  int d = threadIdx.x;
  float v = gelu_f(buf[r * 256 + d]);
  float s1 = v, s2 = v * v;
  int lane = d & 31, wave = d >> 5;
#pragma unroll
  for (int m = 16; m; m >>= 1) {
    s1 += __shfl_xor(s1, m);
    s2 += __shfl_xor(s2, m);
  }
  if (lane == 0) {
    red[wave] = s1;
    red[8 + wave] = s2;
  }
  __syncthreads();
  if (d == 0) {
    float a = 0.f, c = 0.f;
    for (int i = 0; i < 8; ++i) {
      a += red[i];
      c += red[8 + i];
    }
    float mu = a * (1.f / 256.f);
    stats[0] = mu;
    stats[1] = rsqrtf(c * (1.f / 256.f) - mu * mu + 1e-5f);
  }
  __syncthreads();
  buf[r * 256 + d] = g[d] * (v - stats[0]) * stats[1] + b[d];
}

// ---------------- MoE ----------------

__global__ __launch_bounds__(128) void k_gate(const float* f, const float* Wg,
                                              const float* bg, float* gw,
                                              int* gidx, int nv) {
  int wave = threadIdx.x >> 5, lane = threadIdx.x & 31;
  long v = (long)blockIdx.x * 4 + wave;
  if (v >= nv) return;
  float accq[6] = {0.f, 0.f, 0.f, 0.f, 0.f, 0.f};
#pragma unroll
  for (int q = 0; q < 6; ++q) {
    int d = lane + 32 * q;
    float fv = f[v * 192 + d];
#pragma unroll
    for (int e = 0; e < 6; ++e) accq[e] += fv * Wg[d * 6 + e];
  }
#pragma unroll
  for (int e = 0; e < 6; ++e) {
    float s = accq[e];
#pragma unroll
    for (int m = 16; m; m >>= 1) s += __shfl_xor(s, m);
    accq[e] = s;
  }
  if (lane == 0) {
    float lg[6];
#pragma unroll
    for (int e = 0; e < 6; ++e) lg[e] = accq[e] + bg[e];
    int i0 = 0;
    for (int e = 1; e < 6; ++e)
      if (lg[e] > lg[i0]) i0 = e;
    int i1 = -1;
    for (int e = 0; e < 6; ++e)
      if (e != i0 && (i1 < 0 || lg[e] > lg[i1])) i1 = e;
    float e1 = expf(lg[i1] - lg[i0]);
    float s = 1.f + e1;
    gw[v * 2] = 1.f / s;
    gw[v * 2 + 1] = e1 / s;
    gidx[v * 2] = i0;
    gidx[v * 2 + 1] = i1;
  }
}

// experts: oe[v,e] = sum_k gelu(f@We1[e] + be1[e])_k * We2[e,k] + be2[e]
__global__ __launch_bounds__(256) void k_moe(const float* f,
                                             const _Float16* We1P,
                                             const float* be1,
                                             const float* We2,
                                             const float* be2, float* oe,
                                             int M) {
  int wave = threadIdx.x >> 5, lane = threadIdx.x & 31;
  int e = blockIdx.y;
  int mt = (int)blockIdx.x * 8 + wave;
  int row0 = mt * 16;
  if (row0 >= M) return;
  const _Float16* W1 = We1P + (size_t)e * 6 * 192 * 32;  // packed [6][192][32]
  v16h afr[6];
#pragma unroll
  for (int kk = 0; kk < 6; ++kk) afr[kk] = load_a_full(f, 192, row0, kk * 32);
  float rs[8] = {0.f, 0.f, 0.f, 0.f, 0.f, 0.f, 0.f, 0.f};
  for (int nt = 0; nt < 12; ++nt) {
    int c = nt * 16 + (lane & 15);
    v8f acc = splat8(be1[(size_t)e * 192 + c]);
#pragma unroll
    for (int kk = 0; kk < 6; ++kk)
      acc = wmma16(afr[kk], load_b_packed(W1, 192, kk, nt * 16), acc);
    float w2 = We2[(size_t)e * 192 + c];
#pragma unroll
    for (int r = 0; r < 8; ++r) rs[r] += gelu_f(acc[r]) * w2;
  }
#pragma unroll
  for (int r = 0; r < 8; ++r) {
    float s = rs[r];
    s += __shfl_xor(s, 1);
    s += __shfl_xor(s, 2);
    s += __shfl_xor(s, 4);
    s += __shfl_xor(s, 8);
    rs[r] = s;
  }
  if ((lane & 15) == 0) {
    int rbase = row0 + ((lane >> 4) << 3);
    float b2 = be2[e];
#pragma unroll
    for (int r = 0; r < 8; ++r) oe[(size_t)(rbase + r) * 6 + e] = rs[r] + b2;
  }
}

__global__ void k_final(const float* oe, const float* gw, const int* gidx,
                        float* y, int nv) {
  long v = (long)blockIdx.x * blockDim.x + threadIdx.x;
  if (v >= nv) return;
  y[v] = gw[v * 2] * oe[v * 6 + gidx[v * 2]] +
         gw[v * 2 + 1] * oe[v * 6 + gidx[v * 2 + 1]];
}

// ---------------- host ----------------

static inline long cdivl(long a, long b) { return (a + b - 1) / b; }

extern "C" void kernel_launch(void* const* d_in, const int* in_sizes, int n_in,
                              void* d_out, int out_size, void* d_ws,
                              size_t ws_size, hipStream_t stream) {
  (void)n_in;
  (void)out_size;
  (void)ws_size;
  const float* x = (const float*)d_in[0];
  const int* ei = (const int*)d_in[1];
  const float* eattr = (const float*)d_in[2];
  const float* eseq = (const float*)d_in[3];
  const int* veh = (const int*)d_in[4];
  const int* rflat = (const int*)d_in[5];
  const int* rsplit = (const int*)d_in[6];
  const float* g1Wl = (const float*)d_in[7];
  const float* g1bl = (const float*)d_in[8];
  const float* g1Wr = (const float*)d_in[9];
  const float* g1br = (const float*)d_in[10];
  const float* g1We = (const float*)d_in[11];
  const float* g1att = (const float*)d_in[12];
  const float* g1bo = (const float*)d_in[13];
  const float* g2Wl = (const float*)d_in[14];
  const float* g2bl = (const float*)d_in[15];
  const float* g2Wr = (const float*)d_in[16];
  const float* g2br = (const float*)d_in[17];
  const float* g2We = (const float*)d_in[18];
  const float* g2att = (const float*)d_in[19];
  const float* g2bo = (const float*)d_in[20];
  const float* skW = (const float*)d_in[21];
  const float* skb = (const float*)d_in[22];
  const float* ln1g = (const float*)d_in[23];
  const float* ln1b = (const float*)d_in[24];
  const float* ln2g = (const float*)d_in[25];
  const float* ln2b = (const float*)d_in[26];
  const float* tWin = (const float*)d_in[27];
  const float* tbin = (const float*)d_in[28];
  const float* gWih = (const float*)d_in[29];
  const float* gbih = (const float*)d_in[30];
  const float* gWhh = (const float*)d_in[31];
  const float* gbhh = (const float*)d_in[32];
  const float* tWout = (const float*)d_in[33];
  const float* tbout = (const float*)d_in[34];
  const float* cW = (const float*)d_in[35];
  const float* cb = (const float*)d_in[36];
  const float* clng = (const float*)d_in[37];
  const float* clnb = (const float*)d_in[38];
  const float* remb = (const float*)d_in[39];
  const float* fW1 = (const float*)d_in[40];
  const float* fb1 = (const float*)d_in[41];
  const float* flng = (const float*)d_in[42];
  const float* flnb = (const float*)d_in[43];
  const float* fW2 = (const float*)d_in[44];
  const float* fb2 = (const float*)d_in[45];
  const float* mWg = (const float*)d_in[46];
  const float* mbg = (const float*)d_in[47];
  const float* mWe1 = (const float*)d_in[48];
  const float* mbe1 = (const float*)d_in[49];
  const float* mWe2 = (const float*)d_in[50];
  const float* mbe2 = (const float*)d_in[51];

  const int nN = in_sizes[0] / 16;  // 100000
  const int nE = in_sizes[1] / 2;   // 600000
  const int Es = in_sizes[3] / 21;  // 300000
  const int nv = in_sizes[4];       // 50000
  const long tot = (long)nE + nN;

  char* ws = (char*)d_ws;
  const size_t SZ_NODE = (size_t)nN * 128 * 4;
  const size_t OFF_XL = 0;
  const size_t OFF_XR = OFF_XL + SZ_NODE;
  const size_t OFF_ACC = OFF_XR + SZ_NODE;
  const size_t OFF_H = OFF_ACC + SZ_NODE;
  const size_t OFF_LOG = OFF_H + SZ_NODE;
  const size_t OFF_EX = OFF_LOG + (size_t)tot * 4 * 4;
  const size_t OFF_MX = OFF_EX + (size_t)tot * 4 * 4;
  const size_t OFF_DEN = OFF_MX + (size_t)nN * 4 * 4;
  const size_t OFF_SM = OFF_DEN + (size_t)nN * 4 * 4;
  const size_t OFF_OFFS = OFF_SM + 1024;
  const size_t OFF_RZ = OFF_OFFS + (size_t)nv * 4;
  const size_t OFF_PK = OFF_RZ + (size_t)nv * 64 * 4;
  // overlays (regions dead by the time these are written):
  const size_t OFF_Z = OFF_XL;
  const size_t OFF_F1 = OFF_ACC;
  const size_t OFF_F = OFF_H;
  const size_t OFF_OE = OFF_LOG;
  const size_t OFF_GW = OFF_EX;

  float* xl = (float*)(ws + OFF_XL);
  float* xr = (float*)(ws + OFF_XR);
  float* acc = (float*)(ws + OFF_ACC);
  float* h = (float*)(ws + OFF_H);
  float* logits = (float*)(ws + OFF_LOG);
  float* exb = (float*)(ws + OFF_EX);
  unsigned* mx = (unsigned*)(ws + OFF_MX);
  float* den = (float*)(ws + OFF_DEN);
  float* ea_sum = (float*)(ws + OFF_SM);
  float* ctx = (float*)(ws + OFF_SM + 32);
  int* offs = (int*)(ws + OFF_OFFS);
  float* route_z = (float*)(ws + OFF_RZ);
  float* zbuf = (float*)(ws + OFF_Z);
  float* f1 = (float*)(ws + OFF_F1);
  float* fbuf = (float*)(ws + OFF_F);
  float* oe = (float*)(ws + OFF_OE);
  float* gw = (float*)(ws + OFF_GW);
  int* gidx = (int*)(ws + OFF_GW + (size_t)nv * 2 * 4);
  float* yout = (float*)d_out;

  // packed f16 weight arena (half counts)
  _Float16* pk = (_Float16*)(ws + OFF_PK);
  _Float16* pk_g1Wl = pk;
  _Float16* pk_g1Wr = pk_g1Wl + 4096;
  _Float16* pk_g2Wl = pk_g1Wr + 4096;
  _Float16* pk_g2Wr = pk_g2Wl + 16384;
  _Float16* pk_fW1 = pk_g2Wr + 16384;
  _Float16* pk_fW2 = pk_fW1 + 81920;
  _Float16* pk_We1 = pk_fW2 + 49152;
  _Float16* pk_Win = pk_We1 + 221184;
  _Float16* pk_Wih = pk_Win + 4096;
  _Float16* pk_Whh = pk_Wih + 49152;

  auto fill = [&](void* p, unsigned v, long n) {
    k_fill_u32<<<(unsigned)cdivl(n, 256), 256, 0, stream>>>((unsigned*)p, v, n);
  };
  auto packB = [&](const float* B, int K, int N, _Float16* out) {
    long total = (long)((K + 31) >> 5) * N * 32;
    k_packB<<<(unsigned)cdivl(total, 256), 256, 0, stream>>>(B, K, N, out);
  };
  auto gemm = [&](const float* A, int lda, const _Float16* BP,
                  const float* bias, float* C, int ldc, int M, int N, int K) {
    long tiles = (long)(M / 16) * (N / 32);
    k_gemm<<<(unsigned)cdivl(tiles, 8), 256, 0, stream>>>(A, lda, BP, bias, C,
                                                          ldc, M, N, K);
  };

  // --- pack all GEMM weights to f16 fragment-linear layout ---
  packB(g1Wl, 16, 128, pk_g1Wl);
  packB(g1Wr, 16, 128, pk_g1Wr);
  packB(g2Wl, 128, 128, pk_g2Wl);
  packB(g2Wr, 128, 128, pk_g2Wr);
  packB(fW1, 320, 256, pk_fW1);
  packB(fW2, 256, 192, pk_fW2);
  for (int e = 0; e < 6; ++e)
    packB(mWe1 + (size_t)e * 192 * 192, 192, 192,
          pk_We1 + (size_t)e * 6 * 192 * 32);
  packB(tWin, 7, 128, pk_Win);
  k_packH<<<(unsigned)cdivl(384 * 128, 256), 256, 0, stream>>>(gWih, pk_Wih,
                                                               384 * 128);
  k_packH<<<(unsigned)cdivl(384 * 128, 256), 256, 0, stream>>>(gWhh, pk_Whh,
                                                               384 * 128);

  // --- edge_attr mean (self-loop fill) ---
  fill(ea_sum, 0u, 8);
  k_ea_mean<<<(unsigned)cdivl(nE, 256), 256, 0, stream>>>(eattr, ea_sum, nE);

  // --- GAT layer 1 ---
  gemm(x, 16, pk_g1Wl, g1bl, xl, 128, nN, 128, 16);
  gemm(x, 16, pk_g1Wr, g1br, xr, 128, nN, 128, 16);
  fill(mx, 0u, (long)nN * 4);
  fill(den, 0u, (long)nN * 4);
  fill(acc, 0u, (long)nN * 128);
  k_edge_logits<<<(unsigned)cdivl(tot, 4), 128, 0, stream>>>(
      ei, eattr, ea_sum, xl, xr, g1We, g1att, logits, mx, nE, nN);
  k_edge_soft<<<(unsigned)cdivl(tot * 4, 256), 256, 0, stream>>>(
      ei, logits, mx, exb, den, nE, nN);
  k_edge_agg<<<(unsigned)cdivl(tot, 4), 128, 0, stream>>>(ei, exb, den, xl, acc,
                                                          nE, nN);
  k_node1<<<(unsigned)cdivl(nN, 4), 128, 0, stream>>>(acc, g1bo, x, skW, skb,
                                                      ln1g, ln1b, h, nN);

  // --- GAT layer 2 ---
  gemm(h, 128, pk_g2Wl, g2bl, xl, 128, nN, 128, 128);
  gemm(h, 128, pk_g2Wr, g2br, xr, 128, nN, 128, 128);
  fill(mx, 0u, (long)nN * 4);
  fill(den, 0u, (long)nN * 4);
  fill(acc, 0u, (long)nN * 128);
  k_edge_logits<<<(unsigned)cdivl(tot, 4), 128, 0, stream>>>(
      ei, eattr, ea_sum, xl, xr, g2We, g2att, logits, mx, nE, nN);
  k_edge_soft<<<(unsigned)cdivl(tot * 4, 256), 256, 0, stream>>>(
      ei, logits, mx, exb, den, nE, nN);
  k_edge_agg<<<(unsigned)cdivl(tot, 4), 128, 0, stream>>>(ei, exb, den, xl, acc,
                                                          nE, nN);
  k_node2<<<(unsigned)cdivl(nN, 4), 128, 0, stream>>>(acc, g2bo, ln2g, ln2b, h,
                                                      nN);

  // --- fused temporal branch (GRU + edge_to_context -> graph_ctx) ---
  fill(ctx, 0u, 128);
  k_gru<<<(unsigned)cdivl(Es, 64), 128, 0, stream>>>(
      eseq, pk_Win, tbin, pk_Wih, gbih, pk_Whh, gbhh, tWout, tbout, cW, cb,
      clng, clnb, ctx, Es);

  // --- route branch ---
  k_scan<<<1, 1024, 0, stream>>>(rsplit, offs, nv);
  k_route_pool<<<(unsigned)cdivl(nv, 4), 128, 0, stream>>>(remb, rflat, offs,
                                                           rsplit, route_z, nv);

  // --- fusion ---
  k_z<<<(unsigned)cdivl((long)nv * 320, 256), 256, 0, stream>>>(h, veh, ctx,
                                                                route_z, zbuf,
                                                                nv);
  gemm(zbuf, 320, pk_fW1, fb1, f1, 256, nv, 256, 320);
  k_geluln256<<<(unsigned)nv, 256, 0, stream>>>(f1, flng, flnb);
  gemm(f1, 256, pk_fW2, fb2, fbuf, 192, nv, 192, 256);

  // --- MoE head ---
  k_gate<<<(unsigned)cdivl(nv, 4), 128, 0, stream>>>(fbuf, mWg, mbg, gw, gidx,
                                                     nv);
  dim3 mg((unsigned)cdivl(cdivl(nv, 16), 8), 6);
  k_moe<<<mg, 256, 0, stream>>>(fbuf, pk_We1, mbe1, mWe2, mbe2, oe, nv);
  k_final<<<(unsigned)cdivl(nv, 256), 256, 0, stream>>>(oe, gw, gidx, yout, nv);
}
